// VectorQuantizer_7679401525504
// MI455X (gfx1250) — matrix operand order repaired and checked by reading, not compiled, not run on hardware
//
#include <hip/hip_runtime.h>
#include <stdint.h>

typedef float  v2f  __attribute__((ext_vector_type(2)));
typedef float  v4f  __attribute__((ext_vector_type(4)));
typedef float  v8f  __attribute__((ext_vector_type(8)));
typedef __bf16 v8bf  __attribute__((ext_vector_type(8)));
typedef __bf16 v16bf __attribute__((ext_vector_type(16)));

#define DDIM      64
#define KCODES    1024
#define PITCHH    72          // bf16 row pitch (halves): stride 36 dwords -> conflict-free b128
#define BLOCK     512         // 16 wave32 per workgroup
#define WAVES     16
#define ROWS_PER_WAVE  32     // two 16-row A tiles per wave: 2x B-reuse per LDS byte
#define ROWS_PER_BLOCK (WAVES * ROWS_PER_WAVE)

// split x into bf16 hi + bf16 lo (residual); 8 consecutive floats
static __device__ __forceinline__ void splitrow8(const float* __restrict__ p,
                                                 v8bf& hi, v8bf& lo) {
  v4f a = *(const v4f*)p;
  v4f b = *(const v4f*)(p + 4);
  float t[8] = {a.x, a.y, a.z, a.w, b.x, b.y, b.z, b.w};
#pragma unroll
  for (int i = 0; i < 8; ++i) {
    __bf16 h = (__bf16)t[i];
    hi[i] = h;
    lo[i] = (__bf16)(t[i] - (float)h);
  }
}

static __device__ __forceinline__ v16bf cat16(v8bf a, v8bf b) {
  return __builtin_shufflevector(a, b, 0, 1, 2, 3, 4, 5, 6, 7,
                                 8, 9, 10, 11, 12, 13, 14, 15);
}

// split-bf16 VQ: codebook split into bf16 hi/lo planes in LDS (292KB of the
// 320KB WGP LDS), per-wave dual 16x16 tiles, 3-product bf16 WMMA (K=32),
// exact-f32 |w|^2 / gather / losses.
__global__ void __launch_bounds__(BLOCK, 1)
vq_main_kernel(const float* __restrict__ z,
               const float* __restrict__ W,
               float* __restrict__ outZq,
               float* __restrict__ outIdx,
               float* __restrict__ lossAccum) {
  extern __shared__ __bf16 smem[];
  __bf16* Whi = smem;                       // KCODES * PITCHH halves
  __bf16* Wlo = Whi + KCODES * PITCHH;      // KCODES * PITCHH halves
  float*  wn  = (float*)(Wlo + KCODES * PITCHH);  // KCODES floats (|w|^2, exact)

  const int tid  = threadIdx.x;
  const int lane = tid & 31;
  const int l16  = lane & 15;
  const int half = lane >> 4;
  const int wave = tid >> 5;

  // ---- convert codebook (global, L2-hot) -> bf16 hi/lo planes + exact |w|^2 ----
#pragma unroll
  for (int rr = 0; rr < KCODES / BLOCK; ++rr) {
    const int r = tid * (KCODES / BLOCK) + rr;
    const float* wrow = W + (size_t)r * DDIM;
    float s = 0.f;
#pragma unroll
    for (int g = 0; g < DDIM / 8; ++g) {
      v8bf hi, lo;
      splitrow8(wrow + g * 8, hi, lo);
      *(v8bf*)(Whi + r * PITCHH + g * 8) = hi;
      *(v8bf*)(Wlo + r * PITCHH + g * 8) = lo;
      v4f a = *(const v4f*)(wrow + g * 8);
      v4f b = *(const v4f*)(wrow + g * 8 + 4);
      s += a.x * a.x + a.y * a.y + a.z * a.z + a.w * a.w;
      s += b.x * b.x + b.y * b.y + b.z * b.z + b.w * b.w;
    }
    wn[r] = s;
  }

  // ---- two A tiles (32 rows), split into bf16 hi/lo WMMA operands ----
  const int rowBase = (blockIdx.x * WAVES + wave) * ROWS_PER_WAVE;
  const float* zrow0 = z + (size_t)(rowBase + l16) * DDIM;
  const float* zrow1 = zrow0 + 16 * DDIM;
  v16bf ahi0[2], alo0[2], ahi1[2], alo1[2];
#pragma unroll
  for (int c = 0; c < 2; ++c) {
    const int base = c * 32 + half * 8;   // A 16x32 layout: runs [base,+8) & [base+16,+8)
    v8bf h0, l0, h1, l1;
    splitrow8(zrow0 + base, h0, l0);
    splitrow8(zrow0 + base + 16, h1, l1);
    ahi0[c] = cat16(h0, h1); alo0[c] = cat16(l0, l1);
    splitrow8(zrow1 + base, h0, l0);
    splitrow8(zrow1 + base + 16, h1, l1);
    ahi1[c] = cat16(h0, h1); alo1[c] = cat16(l0, l1);
  }

  __syncthreads();

  float minV0[8], minV1[8];
  int   minI0[8], minI1[8];
#pragma unroll
  for (int j = 0; j < 8; ++j) {
    minV0[j] = minV1[j] = 3.402823466e38f;
    minI0[j] = minI1[j] = 0;
  }

  // ---- sweep 64 code tiles; score = |w|^2 - 2*(z.w), 3-product split-bf16 ----
  for (int t = 0; t < KCODES / 16; ++t) {
    const int n0 = t * 16;
    const __bf16* bh = Whi + (n0 + l16) * PITCHH + half * 8;  // B 32x16 layout
    const __bf16* bl = Wlo + (n0 + l16) * PITCHH + half * 8;
    v16bf bhi0 = cat16(*(const v8bf*)(bh),      *(const v8bf*)(bh + 16));
    v16bf bhi1 = cat16(*(const v8bf*)(bh + 32), *(const v8bf*)(bh + 48));
    v16bf blo0 = cat16(*(const v8bf*)(bl),      *(const v8bf*)(bl + 16));
    v16bf blo1 = cat16(*(const v8bf*)(bl + 32), *(const v8bf*)(bl + 48));

    v8f acc0 = {}, acc1 = {};
    acc0 = __builtin_amdgcn_wmma_f32_16x16x32_bf16(false, ahi0[0], false, bhi0, (short)0, acc0, false, false);
    acc1 = __builtin_amdgcn_wmma_f32_16x16x32_bf16(false, ahi1[0], false, bhi0, (short)0, acc1, false, false);
    acc0 = __builtin_amdgcn_wmma_f32_16x16x32_bf16(false, ahi0[1], false, bhi1, (short)0, acc0, false, false);
    acc1 = __builtin_amdgcn_wmma_f32_16x16x32_bf16(false, ahi1[1], false, bhi1, (short)0, acc1, false, false);
    acc0 = __builtin_amdgcn_wmma_f32_16x16x32_bf16(false, ahi0[0], false, blo0, (short)0, acc0, false, false);
    acc1 = __builtin_amdgcn_wmma_f32_16x16x32_bf16(false, ahi1[0], false, blo0, (short)0, acc1, false, false);
    acc0 = __builtin_amdgcn_wmma_f32_16x16x32_bf16(false, ahi0[1], false, blo1, (short)0, acc0, false, false);
    acc1 = __builtin_amdgcn_wmma_f32_16x16x32_bf16(false, ahi1[1], false, blo1, (short)0, acc1, false, false);
    acc0 = __builtin_amdgcn_wmma_f32_16x16x32_bf16(false, alo0[0], false, bhi0, (short)0, acc0, false, false);
    acc1 = __builtin_amdgcn_wmma_f32_16x16x32_bf16(false, alo1[0], false, bhi0, (short)0, acc1, false, false);
    acc0 = __builtin_amdgcn_wmma_f32_16x16x32_bf16(false, alo0[1], false, bhi1, (short)0, acc0, false, false);
    acc1 = __builtin_amdgcn_wmma_f32_16x16x32_bf16(false, alo1[1], false, bhi1, (short)0, acc1, false, false);

    const float wnc  = wn[n0 + l16];
    const int   cand = n0 + l16;
#pragma unroll
    for (int j = 0; j < 8; ++j) {
      float s0 = __builtin_fmaf(-2.f, acc0[j], wnc);
      float s1 = __builtin_fmaf(-2.f, acc1[j], wnc);
      if (s0 < minV0[j]) { minV0[j] = s0; minI0[j] = cand; }
      if (s1 < minV1[j]) { minV1[j] = s1; minI1[j] = cand; }
    }
  }

  // ---- cross-lane argmin within each 16-lane half (rows j / j+8) ----
#pragma unroll
  for (int off = 1; off <= 8; off <<= 1) {
#pragma unroll
    for (int j = 0; j < 8; ++j) {
      float ov; int oi;
      ov = __shfl_xor(minV0[j], off, 32); oi = __shfl_xor(minI0[j], off, 32);
      if (ov < minV0[j] || (ov == minV0[j] && oi < minI0[j])) { minV0[j] = ov; minI0[j] = oi; }
      ov = __shfl_xor(minV1[j], off, 32); oi = __shfl_xor(minI1[j], off, 32);
      if (ov < minV1[j] || (ov == minV1[j] && oi < minI1[j])) { minV1[j] = ov; minI1[j] = oi; }
    }
  }

  if (lane == 0) {
#pragma unroll
    for (int j = 0; j < 8; ++j) {
      outIdx[rowBase + j]      = (float)minI0[j];
      outIdx[rowBase + 16 + j] = (float)minI1[j];
    }
  }
  if (lane == 16) {
#pragma unroll
    for (int j = 0; j < 8; ++j) {
      outIdx[rowBase + 8 + j]  = (float)minI0[j];
      outIdx[rowBase + 24 + j] = (float)minI1[j];
    }
  }

  // ---- gather z_q = exact W[idx] (global, L2-hot), write out, loss ----
  float loss = 0.f;
  const int d0 = lane * 2;
#pragma unroll
  for (int j = 0; j < 8; ++j) {
    const int idx[4] = { __shfl(minI0[j], 0, 32), __shfl(minI0[j], 16, 32),
                         __shfl(minI1[j], 0, 32), __shfl(minI1[j], 16, 32) };
    const int row[4] = { rowBase + j, rowBase + 8 + j,
                         rowBase + 16 + j, rowBase + 24 + j };
#pragma unroll
    for (int q = 0; q < 4; ++q) {
      const size_t off = (size_t)row[q] * DDIM + d0;
      v2f w  = *(const v2f*)(W + (size_t)idx[q] * DDIM + d0);
      v2f zv = *(const v2f*)(z + off);
      *(v2f*)(outZq + off) = w;
      float e1 = zv.x - w.x, e2 = zv.y - w.y;
      loss += e1 * e1 + e2 * e2;
    }
  }
#pragma unroll
  for (int off = 16; off; off >>= 1) loss += __shfl_xor(loss, off, 32);
  if (lane == 0) atomicAdd(lossAccum, loss);
}

__global__ void vq_zero_kernel(float* p) { p[0] = 0.f; }

__global__ void vq_finalize_kernel(const float* __restrict__ s,
                                   float* __restrict__ outLoss, float inv) {
  float m = s[0] * inv;
  outLoss[0] = m;  // commitment loss
  outLoss[1] = m;  // codebook loss (same forward value)
}

extern "C" void kernel_launch(void* const* d_in, const int* in_sizes, int n_in,
                              void* d_out, int out_size, void* d_ws, size_t ws_size,
                              hipStream_t stream) {
  (void)n_in; (void)out_size; (void)ws_size;
  const float* z = (const float*)d_in[0];   // (B, 64)
  const float* W = (const float*)d_in[1];   // (1024, 64)
  const int B = in_sizes[0] / DDIM;

  float* out     = (float*)d_out;
  float* outZq   = out;                          // B*64 floats
  float* outLoss = out + (size_t)B * DDIM;       // 2 floats
  float* outIdx  = outLoss + 2;                  // B floats (indices as floats)
  float* ws      = (float*)d_ws;                 // loss accumulator

  const size_t smemBytes = (size_t)KCODES * PITCHH * 2 * sizeof(__bf16)  // hi+lo
                         + (size_t)KCODES * sizeof(float);               // |w|^2
  hipFuncSetAttribute(reinterpret_cast<const void*>(vq_main_kernel),
                      hipFuncAttributeMaxDynamicSharedMemorySize, (int)smemBytes);

  vq_zero_kernel<<<1, 1, 0, stream>>>(ws);
  vq_main_kernel<<<B / ROWS_PER_BLOCK, BLOCK, smemBytes, stream>>>(
      z, W, outZq, outIdx, ws);
  vq_finalize_kernel<<<1, 1, 0, stream>>>(ws, outLoss,
                                          1.0f / ((float)B * (float)DDIM));
}
